// Attention_79319456023023
// MI455X (gfx1250) — matrix operand order repaired and checked by reading, not compiled, not run on hardware
//
#include <hip/hip_runtime.h>
#include <hip/hip_bf16.h>

// Shapes from the reference
#define H_Q    128
#define HKV    8
#define DH     64
#define RD     32
#define HIDDEN (H_Q * DH)     // 8192
#define KVW    (2 * HKV * DH) // 1024

typedef __attribute__((ext_vector_type(16))) __bf16 v16bf;
typedef __attribute__((ext_vector_type(8)))  __bf16 v8bf;
typedef __attribute__((ext_vector_type(8)))  float  v8f;

union V16 { v16bf v; v8bf h[2]; };

__device__ __forceinline__ unsigned short f32_to_bf16(float f) {
  unsigned int u = __float_as_uint(f);
  unsigned int r = u + 0x7FFFu + ((u >> 16) & 1u); // RNE
  return (unsigned short)(r >> 16);
}
__device__ __forceinline__ float bf16_to_f32(unsigned short s) {
  return __uint_as_float(((unsigned int)s) << 16);
}

// Wave-relative LDS byte offset of a __shared__ pointer: low 32 bits of the
// generic (flat) address are the LDS offset per the CDNA5 aperture mapping.
__device__ __forceinline__ unsigned lds_off(const void* p) {
  return (unsigned)(unsigned long long)p;
}

// CDNA5 async copy: global (SGPR64 base + VGPR32 byte offset) -> LDS, 16B.
// Tracked by ASYNCcnt (not LOADcnt); completion via s_wait_asynccnt.
__device__ __forceinline__ void async_load_b128(const void* gbase,
                                                unsigned goff_bytes,
                                                unsigned lds_byte_off) {
  asm volatile("global_load_async_to_lds_b128 %0, %1, %2"
               :: "v"(lds_byte_off), "v"(goff_bytes), "s"(gbase)
               : "memory");
}
__device__ __forceinline__ void wait_async_le8() {
  asm volatile("s_wait_asynccnt 0x8" ::: "memory");
}
__device__ __forceinline__ void wait_async_0() {
  asm volatile("s_wait_asynccnt 0x0" ::: "memory");
}

// ---------------------------------------------------------------- convert
__global__ void cvt_f32_bf16(const float* __restrict__ in,
                             unsigned short* __restrict__ out, long n) {
  long i = ((long)blockIdx.x * blockDim.x + threadIdx.x) * 8;
  if (i + 8 > n) return;
  float4 a = *(const float4*)(in + i);
  float4 b = *(const float4*)(in + i + 4);
  unsigned short o[8];
  o[0]=f32_to_bf16(a.x); o[1]=f32_to_bf16(a.y); o[2]=f32_to_bf16(a.z); o[3]=f32_to_bf16(a.w);
  o[4]=f32_to_bf16(b.x); o[5]=f32_to_bf16(b.y); o[6]=f32_to_bf16(b.z); o[7]=f32_to_bf16(b.w);
  *(uint4*)(out + i) = *(uint4*)o;
}

// ------------------------------------------------- output store overloads
__device__ __forceinline__ void store_out(float* C, long idx, float v) { C[idx] = v; }
__device__ __forceinline__ void store_out(unsigned short* C, long idx, float v) { C[idx] = f32_to_bf16(v); }

// ---------------------------------------------------------------- NT GEMM
// C[M][N] = A[M][K](bf16 rm) * B[N][K]^T(bf16 rm)
// 128x128 block, BK=64, 256 threads = 8 waves in 4x2 grid; wave tile 32x64
// (2x4 WMMA subtiles -> 1.5 ds_load_b128 per WMMA).
// Double-buffered LDS filled with async global->LDS copies (ASYNCcnt).
template <typename OutT>
__global__ __launch_bounds__(256) void gemm_nt_bf16(
    const unsigned short* __restrict__ A,
    const unsigned short* __restrict__ B,
    OutT* __restrict__ C, int M, int N, int K) {
  constexpr int BM = 128, BN = 128, BK = 64, LDT = 72; // 144B rows: 16B aligned, conflict-free
  __shared__ __align__(16) unsigned short Al[2 * BM * LDT];
  __shared__ __align__(16) unsigned short Bl[2 * BN * LDT];

  const int tid  = threadIdx.x;
  const int lane = tid & 31;
  const int w    = tid >> 5;
  const int m0   = (w & 3) * 32;   // wave row tile
  const int n0   = (w >> 2) * 64;  // wave col tile
  const long mBase = (long)blockIdx.y * BM;
  const long nBase = (long)blockIdx.x * BN;

  const unsigned alBase = lds_off(Al);
  const unsigned blBase = lds_off(Bl);
  constexpr unsigned TILE_B = BM * LDT * 2; // bytes per buffer

  // Per-thread chunk geometry: 1024 16B chunks per tile, 4 per thread.
  int crow[4], ccol[4];
#pragma unroll
  for (int j = 0; j < 4; ++j) {
    int c = tid + j * 256;
    crow[j] = c >> 3;
    ccol[j] = (c & 7) * 8;
  }

  auto issue_tile = [&](int k0, int buf) {
#pragma unroll
    for (int j = 0; j < 4; ++j) {
      unsigned loff = (unsigned)((crow[j] * LDT + ccol[j]) * 2) + buf * TILE_B;
      async_load_b128(A, (unsigned)(((mBase + crow[j]) * (long)K + k0 + ccol[j]) * 2),
                      alBase + loff);
      async_load_b128(B, (unsigned)(((nBase + crow[j]) * (long)K + k0 + ccol[j]) * 2),
                      blBase + loff);
    }
  };

  v8f acc[2][4] = {};

  const int nk = K / BK;
  issue_tile(0, 0);
  for (int ik = 0; ik < nk; ++ik) {
    const int buf = ik & 1;
    if (ik + 1 < nk) {
      issue_tile((ik + 1) * BK, buf ^ 1); // prefetch next tile (8 chunks stay in flight)
      wait_async_le8();                   // oldest 8 (current tile) complete
    } else {
      wait_async_0();
    }
    __syncthreads();

    const unsigned short* Ab = Al + buf * BM * LDT;
    const unsigned short* Bb = Bl + buf * BN * LDT;
#pragma unroll
    for (int ks = 0; ks < 2; ++ks) {
      // A operand (16x32): lane<16 -> K 0..15, lane>=16 -> K 16..31 (contiguous)
      V16 aop[2];
#pragma unroll
      for (int mi = 0; mi < 2; ++mi) {
        int m  = m0 + mi * 16 + (lane & 15);
        int kk = ks * 32 + ((lane >> 4) << 4);
        aop[mi].h[0] = *(const v8bf*)&Ab[m * LDT + kk];
        aop[mi].h[1] = *(const v8bf*)&Ab[m * LDT + kk + 8];
      }
#pragma unroll
      for (int ni = 0; ni < 4; ++ni) {
        // B operand (32x16): lane<16 -> K 0..15, lane>=16 -> K 16..31 (contiguous)
        V16 bop;
        int n  = n0 + ni * 16 + (lane & 15);
        int kk = ks * 32 + ((lane >> 4) << 4);
        bop.h[0] = *(const v8bf*)&Bb[n * LDT + kk];
        bop.h[1] = *(const v8bf*)&Bb[n * LDT + kk + 8];
#pragma unroll
        for (int mi = 0; mi < 2; ++mi)
          acc[mi][ni] = __builtin_amdgcn_wmma_f32_16x16x32_bf16(
              false, aop[mi].v, false, bop.v, (short)0, acc[mi][ni], false, false);
      }
    }
    __syncthreads(); // protect buf being refilled at ik+1
  }

  // Epilogue: C-layout (VGPR r -> row +8*(lane/16)+r, col = lane%16)
  const int rowh = (lane >> 4) << 3;
  const int coln = lane & 15;
#pragma unroll
  for (int mi = 0; mi < 2; ++mi)
#pragma unroll
    for (int ni = 0; ni < 4; ++ni)
#pragma unroll
      for (int r = 0; r < 8; ++r) {
        long row = mBase + m0 + mi * 16 + rowh + r;
        long col = nBase + n0 + ni * 16 + coln;
        store_out(C, row * N + col, acc[mi][ni][r]);
      }
}

// ---------------------------------------------------------------- RoPE
__global__ void rope_kernel(const int* __restrict__ pos,
                            unsigned short* __restrict__ q,
                            unsigned short* __restrict__ kv, int T) {
  long idx = (long)blockIdx.x * blockDim.x + threadIdx.x;
  long total = (long)T * (H_Q + HKV) * (RD / 2);
  if (idx >= total) return;
  int i  = idx & 15;                       // RD/2 = 16
  int hs = (idx >> 4) % (H_Q + HKV);
  int t  = (int)(idx / (16 * (H_Q + HKV)));
  float p = (float)pos[t];
  float freq = p * __expf(-0.5756462732485114f * (float)i); // 10000^(-2i/RD)
  float s, c;
  __sincosf(freq, &s, &c);
  unsigned short* base = (hs < H_Q) ? (q + (long)t * HIDDEN + hs * DH)
                                    : (kv + (long)t * KVW + (hs - H_Q) * DH);
  float x1 = bf16_to_f32(base[i]);
  float x2 = bf16_to_f32(base[i + 16]);
  base[i]      = f32_to_bf16(x1 * c - x2 * s);
  base[i + 16] = f32_to_bf16(x2 * c + x1 * s);
}

// ---------------------------------------------------------------- Flash attention
// Block: 128 queries x 1 head. 8 waves x 16 queries. Key tiles of 32.
__global__ __launch_bounds__(256) void flash_attn(
    const unsigned short* __restrict__ q,   // [T][HIDDEN], rope applied
    const unsigned short* __restrict__ kv,  // [T][KVW] (k at 0, v at HKV*DH)
    unsigned short* __restrict__ out,       // [T][HIDDEN]
    const int* __restrict__ num_seqs, int T) {
  constexpr int QT = 128, KT = 32;
  constexpr int LQ = 72, LK = 72, LV = 40, LP = 40;
  __shared__ __align__(16) unsigned short Ql[QT * LQ];
  __shared__ __align__(16) unsigned short Kl[KT * LK];
  __shared__ __align__(16) unsigned short Vt[DH * LV];      // V transposed: [d][key]
  __shared__ __align__(16) unsigned short Pl[8][16 * LP];   // per-wave probs

  const int tid = threadIdx.x, lane = tid & 31, w = tid >> 5;
  const int h  = blockIdx.y;
  const int kh = h >> 4;                 // G = H/HKV = 16
  const int B  = num_seqs[0];
  const int S  = T / B;
  const int t0 = blockIdx.x * QT;
  const int b  = t0 / S;
  const int s0 = t0 - b * S;

  // Async-load Q tile (128x64)
  const unsigned qlBase = lds_off(Ql);
#pragma unroll
  for (int j = 0; j < 4; ++j) {
    int c = tid + j * 256;
    int row = c >> 3, col = (c & 7) * 8;
    async_load_b128(q, (unsigned)((((long)(t0 + row)) * HIDDEN + h * DH + col) * 2),
                    qlBase + (unsigned)((row * LQ + col) * 2));
  }

  const int qrow = s0 + w * 16;
  v8f Oacc[4] = {};
  float mrow[8], lrow[8];
#pragma unroll
  for (int r = 0; r < 8; ++r) { mrow[r] = -3.0e38f; lrow[r] = 0.f; }

  const int nkt = (s0 + QT - 1) / KT + 1;
  const float scale = 0.125f;
  const unsigned klBase = lds_off(Kl);

  for (int kt = 0; kt < nkt; ++kt) {
    { // K tile async; V tile sync + transposed LDS scatter
      int row = tid >> 3, col = (tid & 7) * 8;
      long src = (long)(b * S + kt * KT + row) * KVW + kh * DH;
      async_load_b128(kv, (unsigned)((src + col) * 2),
                      klBase + (unsigned)((row * LK + col) * 2));
      uint4 vv = *(const uint4*)&kv[src + HKV * DH + col];
      unsigned short vs[8]; *(uint4*)vs = vv;
#pragma unroll
      for (int e = 0; e < 8; ++e) Vt[(col + e) * LV + row] = vs[e];
    }
    wait_async_0();
    __syncthreads();

    // Scores: S[16q x 32k] = Q(16x64) @ K^T
    v8f Sacc[2] = {};
#pragma unroll
    for (int ks = 0; ks < 2; ++ks) {
      V16 aop;
      {
        int m  = w * 16 + (lane & 15);
        int kk = ks * 32 + ((lane >> 4) << 4);
        aop.h[0] = *(const v8bf*)&Ql[m * LQ + kk];
        aop.h[1] = *(const v8bf*)&Ql[m * LQ + kk + 8];
      }
#pragma unroll
      for (int nt = 0; nt < 2; ++nt) {
        V16 bop;
        int n  = nt * 16 + (lane & 15);
        int kk = ks * 32 + ((lane >> 4) << 4);
        bop.h[0] = *(const v8bf*)&Kl[n * LK + kk];
        bop.h[1] = *(const v8bf*)&Kl[n * LK + kk + 8];
        Sacc[nt] = __builtin_amdgcn_wmma_f32_16x16x32_bf16(
            false, aop.v, false, bop.v, (short)0, Sacc[nt], false, false);
      }
    }

    // Causal mask + online softmax (row = qrow + 8*(lane/16) + r)
    const int sqb = qrow + ((lane >> 4) << 3);
    const int sk0 = kt * KT + (lane & 15);
    float pv[2][8];
#pragma unroll
    for (int r = 0; r < 8; ++r) {
      int sq = sqb + r;
      float v0 = Sacc[0][r] * scale;
      float v1 = Sacc[1][r] * scale;
      if (sk0 > sq)      v0 = -3.0e38f;
      if (sk0 + 16 > sq) v1 = -3.0e38f;
      float t = fmaxf(v0, v1);
#pragma unroll
      for (int msk = 8; msk >= 1; msk >>= 1) t = fmaxf(t, __shfl_xor(t, msk, 32));
      float mnew  = fmaxf(mrow[r], t);
      float alpha = __expf(mrow[r] - mnew);
      float p0 = __expf(v0 - mnew);
      float p1 = __expf(v1 - mnew);
      float sum = p0 + p1;
#pragma unroll
      for (int msk = 8; msk >= 1; msk >>= 1) sum += __shfl_xor(sum, msk, 32);
      lrow[r] = lrow[r] * alpha + sum;
      mrow[r] = mnew;
      pv[0][r] = p0; pv[1][r] = p1;
#pragma unroll
      for (int ot = 0; ot < 4; ++ot) Oacc[ot][r] *= alpha;
    }

    // P (C-layout) -> per-wave LDS -> A-operand layout
    unsigned short* pw = Pl[w];
    const int prow = (lane >> 4) << 3;
#pragma unroll
    for (int nt = 0; nt < 2; ++nt)
#pragma unroll
      for (int r = 0; r < 8; ++r)
        pw[(prow + r) * LP + nt * 16 + (lane & 15)] = f32_to_bf16(pv[nt][r]);

    // O(16x64) += P(16x32) @ V(32x64); V^T gives contiguous-K B operand
    V16 aop;
    {
      int m  = lane & 15;
      int kk = (lane >> 4) << 4;
      aop.h[0] = *(const v8bf*)&pw[m * LP + kk];
      aop.h[1] = *(const v8bf*)&pw[m * LP + kk + 8];
    }
#pragma unroll
    for (int ot = 0; ot < 4; ++ot) {
      V16 bop;
      int d  = ot * 16 + (lane & 15);
      int kk = (lane >> 4) << 4;
      bop.h[0] = *(const v8bf*)&Vt[d * LV + kk];
      bop.h[1] = *(const v8bf*)&Vt[d * LV + kk + 8];
      Oacc[ot] = __builtin_amdgcn_wmma_f32_16x16x32_bf16(
          false, aop.v, false, bop.v, (short)0, Oacc[ot], false, false);
    }
    __syncthreads();
  }

  // Normalize and store
  const int rowh = (lane >> 4) << 3;
#pragma unroll
  for (int ot = 0; ot < 4; ++ot)
#pragma unroll
    for (int r = 0; r < 8; ++r) {
      float v = Oacc[ot][r] / lrow[r];
      long tq = (long)t0 + w * 16 + rowh + r;
      out[tq * HIDDEN + h * DH + ot * 16 + (lane & 15)] = f32_to_bf16(v);
    }
}

// ---------------------------------------------------------------- launch
extern "C" void kernel_launch(void* const* d_in, const int* in_sizes, int n_in,
                              void* d_out, int out_size, void* d_ws, size_t ws_size,
                              hipStream_t stream) {
  const int*   pos    = (const int*)d_in[0];
  const float* hidden = (const float*)d_in[1];
  const float* Wq     = (const float*)d_in[2];
  const float* Wkv    = (const float*)d_in[3];
  const float* Wo     = (const float*)d_in[4];
  const int*   nseq   = (const int*)d_in[5];
  const int T = in_sizes[1] / HIDDEN;   // 2048

  char* ws = (char*)d_ws;
  unsigned short* h_bf    = (unsigned short*)ws; ws += (size_t)T * HIDDEN * 2;
  unsigned short* wq_bf   = (unsigned short*)ws; ws += (size_t)HIDDEN * HIDDEN * 2;
  unsigned short* wkv_bf  = (unsigned short*)ws; ws += (size_t)KVW * HIDDEN * 2;
  unsigned short* wo_bf   = (unsigned short*)ws; ws += (size_t)HIDDEN * HIDDEN * 2;
  unsigned short* q_bf    = (unsigned short*)ws; ws += (size_t)T * HIDDEN * 2;
  unsigned short* kv_bf   = (unsigned short*)ws; ws += (size_t)T * KVW * 2;
  unsigned short* attn_bf = (unsigned short*)ws; ws += (size_t)T * HIDDEN * 2;

  auto cvt = [&](const float* in, unsigned short* out, long n) {
    long blocks = (n / 8 + 255) / 256;
    cvt_f32_bf16<<<dim3((unsigned)blocks), 256, 0, stream>>>(in, out, n);
  };
  cvt(hidden, h_bf,  (long)T * HIDDEN);
  cvt(Wq,     wq_bf, (long)HIDDEN * HIDDEN);
  cvt(Wkv,    wkv_bf,(long)KVW * HIDDEN);
  cvt(Wo,     wo_bf, (long)HIDDEN * HIDDEN);

  gemm_nt_bf16<unsigned short><<<dim3(HIDDEN / 128, T / 128), 256, 0, stream>>>(
      h_bf, wq_bf, q_bf, T, HIDDEN, HIDDEN);
  gemm_nt_bf16<unsigned short><<<dim3(KVW / 128, T / 128), 256, 0, stream>>>(
      h_bf, wkv_bf, kv_bf, T, KVW, HIDDEN);

  {
    long total = (long)T * (H_Q + HKV) * (RD / 2);
    rope_kernel<<<dim3((unsigned)((total + 255) / 256)), 256, 0, stream>>>(pos, q_bf, kv_bf, T);
  }

  flash_attn<<<dim3(T / 128, H_Q), 256, 0, stream>>>(q_bf, kv_bf, attn_bf, nseq, T);

  gemm_nt_bf16<float><<<dim3(HIDDEN / 128, T / 128), 256, 0, stream>>>(
      attn_bf, wo_bf, (float*)d_out, T, HIDDEN, HIDDEN);
}
